// DecoderOnlyTransformer_20641612824850
// MI455X (gfx1250) — compile-verified, matrix operand order checked by reading, not run on hardware
//
#include <hip/hip_runtime.h>

#define V_  50257
#define S_  2048
#define D_  768
#define NL  4
#define H_  12
#define HD_ 64
#define DF_ 3072
#define B_  2
#define TOK (B_*S_)   // 4096 tokens

typedef __attribute__((ext_vector_type(16))) __bf16 bf16x16;
typedef __attribute__((ext_vector_type(8)))  float  floatx8;
typedef unsigned short u16;

union FragB { bf16x16 v; uint4 q[2]; };

#ifndef __has_builtin
#define __has_builtin(x) 0
#endif

#if __has_builtin(__builtin_amdgcn_global_load_async_to_lds_b128)
#define HAVE_ASYNC_LDS 1
#else
#define HAVE_ASYNC_LDS 0
#endif

#if HAVE_ASYNC_LDS
typedef int v4i_gcc __attribute__((vector_size(16)));
__device__ __forceinline__ void gld_async_b128(const void* g, void* l) {
  __builtin_amdgcn_global_load_async_to_lds_b128(
      (__attribute__((address_space(1))) v4i_gcc*)g,
      (__attribute__((address_space(3))) v4i_gcc*)l, 0, 0);
}
#if __has_builtin(__builtin_amdgcn_s_wait_asynccnt)
#define WAIT_ASYNC(n) __builtin_amdgcn_s_wait_asynccnt(n)
#else
#define WAIT_ASYNC(n) asm volatile("s_wait_asynccnt %0" :: "i"(n) : "memory")
#endif
#endif

__device__ __forceinline__ u16 f2bf(float f) {
  unsigned u = __float_as_uint(f);
  unsigned r = u + 0x7FFFu + ((u >> 16) & 1u);   // round-to-nearest-even
  return (u16)(r >> 16);
}

__device__ __forceinline__ floatx8 wmma_bf16(bf16x16 a, bf16x16 b, floatx8 c) {
  return __builtin_amdgcn_wmma_f32_16x16x32_bf16(false, a, false, b, (short)0, c,
                                                 false, false);
}

// ---------------------------------------------------------------------------
// Embedding + sinusoidal positional encoding, dual f32/bf16 output
// ---------------------------------------------------------------------------
__global__ __launch_bounds__(256)
void embed_kernel(const int* __restrict__ x, const float* __restrict__ tok,
                  float* __restrict__ h, u16* __restrict__ hb)
{
  int bs = blockIdx.x;
  int pos = bs % S_;
  int token = x[bs];
#pragma unroll
  for (int j = 0; j < 3; j++) {
    int d = threadIdx.x + j * 256;
    float freq = __expf(-9.210340371976184f * (float)(d & ~1) / (float)D_);
    float ang = (float)pos * freq;
    float pe = (d & 1) ? __cosf(ang) : __sinf(ang);
    float val = tok[(size_t)token * D_ + d] + pe;
    h[(size_t)bs * D_ + d] = val;
    hb[(size_t)bs * D_ + d] = f2bf(val);
  }
}

// ---------------------------------------------------------------------------
// f32 [rows,cols] -> bf16 transposed [cols,rows]
// ---------------------------------------------------------------------------
__global__ __launch_bounds__(256)
void transpose_bf16_kernel(const float* __restrict__ src, u16* __restrict__ dst,
                           int rows, int cols)
{
  size_t idx = (size_t)blockIdx.x * 256 + threadIdx.x;
  if (idx >= (size_t)rows * cols) return;
  int r = (int)(idx / cols);
  int c = (int)(idx % cols);
  dst[(size_t)c * rows + r] = f2bf(src[idx]);
}

// ---------------------------------------------------------------------------
// out = LayerNorm(x (+ y)) * g + b  -> f32 and bf16 copies (y may be null)
// ---------------------------------------------------------------------------
__global__ __launch_bounds__(256)
void add_ln_kernel(const float* __restrict__ x, const float* __restrict__ y,
                   const float* __restrict__ g, const float* __restrict__ bparam,
                   float* __restrict__ outf, u16* __restrict__ outb)
{
  __shared__ float s1[256], s2[256];
  int t = threadIdx.x, row = blockIdx.x;
  const float* xr = x + (size_t)row * D_;
  const float* yr = y ? (y + (size_t)row * D_) : nullptr;
  float v[3];
  float sum = 0.f, sq = 0.f;
#pragma unroll
  for (int j = 0; j < 3; j++) {
    int d = t + j * 256;
    float val = xr[d] + (yr ? yr[d] : 0.f);
    v[j] = val; sum += val; sq += val * val;
  }
  s1[t] = sum; s2[t] = sq;
  __syncthreads();
  for (int off = 128; off > 0; off >>= 1) {
    if (t < off) { s1[t] += s1[t + off]; s2[t] += s2[t + off]; }
    __syncthreads();
  }
  float mean = s1[0] * (1.f / (float)D_);
  float var  = s2[0] * (1.f / (float)D_) - mean * mean;
  float inv  = rsqrtf(var + 1e-5f);
#pragma unroll
  for (int j = 0; j < 3; j++) {
    int d = t + j * 256;
    float o = (v[j] - mean) * inv * g[d] + bparam[d];
    outf[(size_t)row * D_ + d] = o;
    outb[(size_t)row * D_ + d] = f2bf(o);
  }
}

// ---------------------------------------------------------------------------
// C[M,N] = A[M,K](bf16) * Bt[N,K](bf16) (+bias)(+relu), double-buffered LDS,
// async global->LDS copies when available.  Block tile 128x128, 4 waves,
// each wave 64x64 via 4x4 wmma accumulators.  M%128==0, K%32==0; N guarded.
// ---------------------------------------------------------------------------
#define LDT 40               // LDS row stride in elems (32+8 pad), 80B
#define TILE_ELEMS (128*LDT)

__device__ __forceinline__ void stage_gemm_tile(
    const u16* __restrict__ A, const u16* __restrict__ Bt,
    u16* As, u16* Bs, int m0, int n0, int Nn, int K, int k0, int tid)
{
#pragma unroll
  for (int i = 0; i < 4; i++) {
    int chunk = tid + i * 128;
    int r  = chunk >> 2;
    int cc = (chunk & 3) * 8;
    int nrow = n0 + r; if (nrow > Nn - 1) nrow = Nn - 1;   // clamp edge rows
    const u16* ga = &A[(size_t)(m0 + r) * K + k0 + cc];
    const u16* gb = &Bt[(size_t)nrow * K + k0 + cc];
#if HAVE_ASYNC_LDS
    gld_async_b128(ga, &As[r * LDT + cc]);
    gld_async_b128(gb, &Bs[r * LDT + cc]);
#else
    *reinterpret_cast<uint4*>(&As[r * LDT + cc]) =
        *reinterpret_cast<const uint4*>(ga);
    *reinterpret_cast<uint4*>(&Bs[r * LDT + cc]) =
        *reinterpret_cast<const uint4*>(gb);
#endif
  }
}

__device__ __forceinline__ void gemm_compute_tile(
    const u16* As, const u16* Bs, floatx8 (&acc)[4][4],
    int wm, int wn, int lrow, int kb8)
{
  FragB af[4], bf[4];
#pragma unroll
  for (int fm = 0; fm < 4; fm++) {
    int r = wm * 64 + fm * 16 + lrow;
    af[fm].q[0] = *reinterpret_cast<const uint4*>(&As[r * LDT + kb8]);
    af[fm].q[1] = *reinterpret_cast<const uint4*>(&As[r * LDT + kb8 + 16]);
  }
#pragma unroll
  for (int fn = 0; fn < 4; fn++) {
    int r = wn * 64 + fn * 16 + lrow;
    bf[fn].q[0] = *reinterpret_cast<const uint4*>(&Bs[r * LDT + kb8]);
    bf[fn].q[1] = *reinterpret_cast<const uint4*>(&Bs[r * LDT + kb8 + 16]);
  }
#pragma unroll
  for (int fm = 0; fm < 4; fm++)
#pragma unroll
    for (int fn = 0; fn < 4; fn++)
      acc[fm][fn] = wmma_bf16(af[fm].v, bf[fn].v, acc[fm][fn]);
}

__global__ __launch_bounds__(128)
void gemm_bf16_kernel(const u16* __restrict__ A, const u16* __restrict__ Bt,
                      const float* __restrict__ bias,
                      float* __restrict__ outf, u16* __restrict__ outb,
                      int M, int Nn, int K, int relu)
{
  __shared__ u16 As[2][TILE_ELEMS];
  __shared__ u16 Bs[2][TILE_ELEMS];
  const int tid  = threadIdx.x;
  const int lane = tid & 31;
  const int wid  = tid >> 5;
  const int wm   = wid >> 1;
  const int wn   = wid & 1;
  const int m0 = blockIdx.y * 128;
  const int n0 = blockIdx.x * 128;
  const int lrow = lane & 15;
  const int kb8  = (lane >> 4) * 8;

  floatx8 acc[4][4];
#pragma unroll
  for (int i = 0; i < 4; i++)
#pragma unroll
    for (int j = 0; j < 4; j++)
#pragma unroll
      for (int e = 0; e < 8; e++) acc[i][j][e] = 0.f;

#if HAVE_ASYNC_LDS
  // async pipeline: keep two tile-groups (8 loads/thread each) in flight
  stage_gemm_tile(A, Bt, As[0], Bs[0], m0, n0, Nn, K, 0, tid);
  if (32 < K) stage_gemm_tile(A, Bt, As[1], Bs[1], m0, n0, Nn, K, 32, tid);
  int buf = 0;
  for (int k0 = 0; k0 < K; k0 += 32, buf ^= 1) {
    if (k0 + 32 < K) { WAIT_ASYNC(8); } else { WAIT_ASYNC(0); }
    __syncthreads();
    gemm_compute_tile(As[buf], Bs[buf], acc, wm, wn, lrow, kb8);
    __syncthreads();
    if (k0 + 64 < K)
      stage_gemm_tile(A, Bt, As[buf], Bs[buf], m0, n0, Nn, K, k0 + 64, tid);
  }
#else
  // register-staged double buffer: stage next tile before computing current
  stage_gemm_tile(A, Bt, As[0], Bs[0], m0, n0, Nn, K, 0, tid);
  int buf = 0;
  for (int k0 = 0; k0 < K; k0 += 32, buf ^= 1) {
    __syncthreads();
    if (k0 + 32 < K) {
      __builtin_prefetch(&A[(size_t)(m0 + tid) * K + k0 + 64], 0, 1);
      stage_gemm_tile(A, Bt, As[buf ^ 1], Bs[buf ^ 1], m0, n0, Nn, K, k0 + 32, tid);
    }
    gemm_compute_tile(As[buf], Bs[buf], acc, wm, wn, lrow, kb8);
  }
#endif

  const int roff = (lane >> 4) * 8;   // lanes 16..31 hold rows 8..15
#pragma unroll
  for (int fm = 0; fm < 4; fm++) {
#pragma unroll
    for (int fn = 0; fn < 4; fn++) {
      int col = n0 + wn * 64 + fn * 16 + lrow;
      if (col >= Nn) continue;
      float badd = bias ? bias[col] : 0.f;
#pragma unroll
      for (int e = 0; e < 8; e++) {
        int row = m0 + wm * 64 + fm * 16 + roff + e;
        float val = acc[fm][fn][e] + badd;
        if (relu) val = fmaxf(val, 0.f);
        if (outb) outb[(size_t)row * Nn + col] = f2bf(val);
        else      outf[(size_t)row * Nn + col] = val;
      }
    }
  }
}

// ---------------------------------------------------------------------------
// Flash attention: block = 64 queries (4 waves x 16 rows), causal, HD=64.
// q/k/v/o are bf16 [B,S,H*HD] token-major; scale = 1/sqrt(64).
// ---------------------------------------------------------------------------
#define LVT 72   // LDS stride (64 + 8 pad), 144B = 16B-aligned
__global__ __launch_bounds__(128)
void attention_kernel(const u16* __restrict__ q, const u16* __restrict__ k,
                      const u16* __restrict__ v, u16* __restrict__ o)
{
  __shared__ u16 Ksh[64 * LVT];          // [key][hd]
  __shared__ u16 Vsh[64 * LVT];          // [hd][key]  (V transposed)
  __shared__ u16 Psh[4 * 16 * LVT];      // per-wave P tile [qrow][key]
  const int tid  = threadIdx.x;
  const int lane = tid & 31;
  const int w    = tid >> 5;
  const int bh   = blockIdx.y;
  const int bb   = bh / H_, hh = bh % H_;
  const int qt   = blockIdx.x;
  const int lrow = lane & 15;
  const int kb8  = (lane >> 4) * 8;
  const int roff = (lane >> 4) * 8;
  const int q0w  = qt * 64 + w * 16;

  FragB qf[2];
  {
    size_t rowoff = ((size_t)(bb * S_ + q0w + lrow)) * D_ + hh * HD_;
#pragma unroll
    for (int ks = 0; ks < 2; ks++) {
      qf[ks].q[0] = *reinterpret_cast<const uint4*>(&q[rowoff + ks * 32 + kb8]);
      qf[ks].q[1] = *reinterpret_cast<const uint4*>(&q[rowoff + ks * 32 + kb8 + 16]);
    }
  }

  float m_run[8], l_run[8];
  floatx8 oacc[4];
#pragma unroll
  for (int e = 0; e < 8; e++) { m_run[e] = -3.0e38f; l_run[e] = 0.f; }
#pragma unroll
  for (int nt = 0; nt < 4; nt++)
#pragma unroll
    for (int e = 0; e < 8; e++) oacc[nt][e] = 0.f;

  for (int kt = 0; kt <= qt; kt++) {
    const int kbase = kt * 64;
#pragma unroll
    for (int i = 0; i < 4; i++) {          // stage K (async) and V^T (sync)
      int chunk = tid + i * 128;
      int r  = chunk >> 3;
      int cc = (chunk & 7) * 8;
      size_t goff = ((size_t)(bb * S_ + kbase + r)) * D_ + hh * HD_ + cc;
#if HAVE_ASYNC_LDS
      gld_async_b128(&k[goff], &Ksh[r * LVT + cc]);
#else
      *reinterpret_cast<uint4*>(&Ksh[r * LVT + cc]) =
          *reinterpret_cast<const uint4*>(&k[goff]);
#endif
      union { uint4 u; u16 s[8]; } vv;
      vv.u = *reinterpret_cast<const uint4*>(&v[goff]);
#pragma unroll
      for (int j = 0; j < 8; j++) Vsh[(cc + j) * LVT + r] = vv.s[j];
    }
#if HAVE_ASYNC_LDS
    WAIT_ASYNC(0);
#endif
    __syncthreads();

    // scores = Q * K^T  (K tile rows are already the "Bt" layout)
    floatx8 sacc[4];
#pragma unroll
    for (int nt = 0; nt < 4; nt++)
#pragma unroll
      for (int e = 0; e < 8; e++) sacc[nt][e] = 0.f;
#pragma unroll
    for (int ks = 0; ks < 2; ks++) {
#pragma unroll
      for (int nt = 0; nt < 4; nt++) {
        FragB kf;
        int rr = nt * 16 + lrow;
        kf.q[0] = *reinterpret_cast<const uint4*>(&Ksh[rr * LVT + ks * 32 + kb8]);
        kf.q[1] = *reinterpret_cast<const uint4*>(&Ksh[rr * LVT + ks * 32 + kb8 + 16]);
        sacc[nt] = wmma_bf16(qf[ks].v, kf.v, sacc[nt]);
      }
    }

    // scale + causal mask
    float sc[4][8];
#pragma unroll
    for (int nt = 0; nt < 4; nt++)
#pragma unroll
      for (int e = 0; e < 8; e++) {
        int row = q0w + roff + e;
        int col = kbase + nt * 16 + lrow;
        float s = sacc[nt][e] * 0.125f;
        sc[nt][e] = (col > row) ? -3.0e38f : s;
      }

    // online softmax: row reductions across the 16 lanes sharing each row
#pragma unroll
    for (int e = 0; e < 8; e++) {
      float rmax = fmaxf(fmaxf(sc[0][e], sc[1][e]), fmaxf(sc[2][e], sc[3][e]));
#pragma unroll
      for (int off = 8; off > 0; off >>= 1)
        rmax = fmaxf(rmax, __shfl_xor(rmax, off, 32));
      float mnew = fmaxf(m_run[e], rmax);
      float corr = __expf(m_run[e] - mnew);
      float rsum = 0.f;
#pragma unroll
      for (int nt = 0; nt < 4; nt++) {
        float p = __expf(sc[nt][e] - mnew);
        sc[nt][e] = p;
        rsum += p;
      }
#pragma unroll
      for (int off = 8; off > 0; off >>= 1)
        rsum += __shfl_xor(rsum, off, 32);
      l_run[e] = l_run[e] * corr + rsum;
      m_run[e] = mnew;
#pragma unroll
      for (int nt = 0; nt < 4; nt++) oacc[nt][e] *= corr;
    }

    // transpose P (D-layout -> A-layout) through this wave's LDS slice
#pragma unroll
    for (int nt = 0; nt < 4; nt++)
#pragma unroll
      for (int e = 0; e < 8; e++)
        Psh[(w * 16 + roff + e) * LVT + nt * 16 + lrow] = f2bf(sc[nt][e]);

    // O += P * V   (DS ops of a wave are in-order: stores above precede loads)
#pragma unroll
    for (int ks = 0; ks < 2; ks++) {
      FragB pf;
      pf.q[0] = *reinterpret_cast<const uint4*>(&Psh[(w * 16 + lrow) * LVT + ks * 32 + kb8]);
      pf.q[1] = *reinterpret_cast<const uint4*>(&Psh[(w * 16 + lrow) * LVT + ks * 32 + kb8 + 16]);
#pragma unroll
      for (int nt = 0; nt < 4; nt++) {
        FragB vf;
        int rr = nt * 16 + lrow;
        vf.q[0] = *reinterpret_cast<const uint4*>(&Vsh[rr * LVT + ks * 32 + kb8]);
        vf.q[1] = *reinterpret_cast<const uint4*>(&Vsh[rr * LVT + ks * 32 + kb8 + 16]);
        oacc[nt] = wmma_bf16(pf.v, vf.v, oacc[nt]);
      }
    }
    __syncthreads();
  }

#pragma unroll
  for (int nt = 0; nt < 4; nt++)
#pragma unroll
    for (int e = 0; e < 8; e++) {
      int row = q0w + roff + e;
      int col = hh * HD_ + nt * 16 + lrow;
      o[((size_t)(bb * S_ + row)) * D_ + col] = f2bf(oacc[nt][e] / l_run[e]);
    }
}

// ---------------------------------------------------------------------------
// host
// ---------------------------------------------------------------------------
static inline int cdiv(int a, int b) { return (a + b - 1) / b; }

static void launch_gemm(hipStream_t st, const u16* A, const u16* Bt,
                        const float* bias, float* outf, u16* outb,
                        int M, int Nn, int K, int relu)
{
  dim3 grid(cdiv(Nn, 128), cdiv(M, 128));
  gemm_bf16_kernel<<<grid, dim3(128), 0, st>>>(A, Bt, bias, outf, outb, M, Nn, K, relu);
}

extern "C" void kernel_launch(void* const* d_in, const int* in_sizes, int n_in,
                              void* d_out, int out_size, void* d_ws, size_t ws_size,
                              hipStream_t stream)
{
  (void)in_sizes; (void)n_in; (void)out_size; (void)ws_size;
  const int*   x       = (const int*)  d_in[0];
  const float* tok_emb = (const float*)d_in[1];
  const float* wq      = (const float*)d_in[2];
  const float* wk      = (const float*)d_in[3];
  const float* wv      = (const float*)d_in[4];
  const float* wo      = (const float*)d_in[5];
  const float* w1      = (const float*)d_in[6];
  const float* b1      = (const float*)d_in[7];
  const float* w2      = (const float*)d_in[8];
  const float* b2      = (const float*)d_in[9];
  const float* ln1_g   = (const float*)d_in[10];
  const float* ln1_b   = (const float*)d_in[11];
  const float* ln2_g   = (const float*)d_in[12];
  const float* ln2_b   = (const float*)d_in[13];
  const float* lnf_g   = (const float*)d_in[14];
  const float* lnf_b   = (const float*)d_in[15];
  const float* lm_w    = (const float*)d_in[16];
  const float* lm_b    = (const float*)d_in[17];
  float* logits = (float*)d_out;

  char* cur = (char*)d_ws;
  auto carve = [&](size_t bytes) -> void* {
    void* p = (void*)cur;
    cur += (bytes + 255) & ~((size_t)255);
    return p;
  };
  u16*   WQt  = (u16*)carve((size_t)NL * D_ * D_ * 2);
  u16*   WKt  = (u16*)carve((size_t)NL * D_ * D_ * 2);
  u16*   WVt  = (u16*)carve((size_t)NL * D_ * D_ * 2);
  u16*   WOt  = (u16*)carve((size_t)NL * D_ * D_ * 2);
  u16*   W1t  = (u16*)carve((size_t)NL * D_ * DF_ * 2);
  u16*   W2t  = (u16*)carve((size_t)NL * D_ * DF_ * 2);
  u16*   LMWt = (u16*)carve((size_t)V_ * D_ * 2);
  float* hbuf = (float*)carve((size_t)TOK * D_ * 4);
  u16*   hb   = (u16*)carve((size_t)TOK * D_ * 2);
  u16*   qb   = (u16*)carve((size_t)TOK * D_ * 2);
  u16*   kbuf = (u16*)carve((size_t)TOK * D_ * 2);
  u16*   vbuf = (u16*)carve((size_t)TOK * D_ * 2);
  u16*   ob   = (u16*)carve((size_t)TOK * D_ * 2);
  float* tmp  = (float*)carve((size_t)TOK * D_ * 4);
  u16*   ff1b = (u16*)carve((size_t)TOK * DF_ * 2);

  // --- weight conversion: f32 -> bf16, transposed to [N,K] row-major ---
  for (int i = 0; i < NL; i++) {
    for (int hh = 0; hh < H_; hh++) {
      size_t so   = ((size_t)i * H_ + hh) * D_ * HD_;        // [D,HD] per head
      size_t dofs = (size_t)i * D_ * D_ + (size_t)hh * HD_ * D_;
      int g = cdiv(D_ * HD_, 256);
      transpose_bf16_kernel<<<g, 256, 0, stream>>>(wq + so, WQt + dofs, D_, HD_);
      transpose_bf16_kernel<<<g, 256, 0, stream>>>(wk + so, WKt + dofs, D_, HD_);
      transpose_bf16_kernel<<<g, 256, 0, stream>>>(wv + so, WVt + dofs, D_, HD_);
    }
    transpose_bf16_kernel<<<cdiv(D_ * D_, 256), 256, 0, stream>>>(
        wo + (size_t)i * D_ * D_, WOt + (size_t)i * D_ * D_, D_, D_);
    transpose_bf16_kernel<<<cdiv(D_ * DF_, 256), 256, 0, stream>>>(
        w1 + (size_t)i * D_ * DF_, W1t + (size_t)i * D_ * DF_, D_, DF_);
    transpose_bf16_kernel<<<cdiv(D_ * DF_, 256), 256, 0, stream>>>(
        w2 + (size_t)i * D_ * DF_, W2t + (size_t)i * D_ * DF_, DF_, D_);
  }
  {
    size_t total = (size_t)D_ * V_;
    transpose_bf16_kernel<<<(int)((total + 255) / 256), 256, 0, stream>>>(
        lm_w, LMWt, D_, V_);
  }

  // --- forward pass ---
  embed_kernel<<<TOK, 256, 0, stream>>>(x, tok_emb, hbuf, hb);

  for (int i = 0; i < NL; i++) {
    launch_gemm(stream, hb, WQt + (size_t)i * D_ * D_, nullptr, nullptr, qb,   TOK, D_, D_, 0);
    launch_gemm(stream, hb, WKt + (size_t)i * D_ * D_, nullptr, nullptr, kbuf, TOK, D_, D_, 0);
    launch_gemm(stream, hb, WVt + (size_t)i * D_ * D_, nullptr, nullptr, vbuf, TOK, D_, D_, 0);
    attention_kernel<<<dim3(S_ / 64, B_ * H_), 128, 0, stream>>>(qb, kbuf, vbuf, ob);
    launch_gemm(stream, ob, WOt + (size_t)i * D_ * D_, nullptr, tmp, nullptr, TOK, D_, D_, 0);
    add_ln_kernel<<<TOK, 256, 0, stream>>>(hbuf, tmp, ln1_g + (size_t)i * D_,
                                           ln1_b + (size_t)i * D_, hbuf, hb);
    launch_gemm(stream, hb, W1t + (size_t)i * D_ * DF_, b1 + (size_t)i * DF_,
                nullptr, ff1b, TOK, DF_, D_, 1);
    launch_gemm(stream, ff1b, W2t + (size_t)i * D_ * DF_, b2 + (size_t)i * D_,
                tmp, nullptr, TOK, D_, DF_, 0);
    add_ln_kernel<<<TOK, 256, 0, stream>>>(hbuf, tmp, ln2_g + (size_t)i * D_,
                                           ln2_b + (size_t)i * D_, hbuf, hb);
  }
  add_ln_kernel<<<TOK, 256, 0, stream>>>(hbuf, nullptr, lnf_g, lnf_b, hbuf, hb);
  launch_gemm(stream, hb, LMWt, lm_b, logits, nullptr, TOK, V_, D_, 0);
}
